// OuterProductLayer_15762529976838
// MI455X (gfx1250) — matrix operand order repaired and checked by reading, not compiled, not run on hardware
//
#include <hip/hip_runtime.h>

// out[b,p] = sum_{a,c} x[b,ROW[p],c] * w[a,p,c] * x[b,COL[p],a]
// Per pair p: T = Xrow_tile(16x32) @ W_p^T via v_wmma_f32_16x16x32_f16 (fp32 acc),
// then u = T .* Xcol and a DS-free reduction tree: xor1/xor2 via DPP16 quad_perm,
// xor4 via DPP8, xor8 via v_permlane16; register merges are single v_cndmask ops
// against compile-time SGPR wave masks.

namespace {
constexpr int kF = 39;
constexpr int kK = 32;                         // inner dims a and c
constexpr int kP = kF * (kF - 1) / 2;          // 741 pairs
constexpr int kTilesPerWave = 8;               // 16-batch tiles per wave (reuse W_p)
constexpr int kWavesPerBlock = 8;              // 256 threads
constexpr int kTileCount = 8192 / 16;          // 512 batch tiles
constexpr int kUnitsPerPair = kTileCount / kTilesPerWave;  // 64
}

typedef __attribute__((ext_vector_type(16))) _Float16 v16h;
typedef __attribute__((ext_vector_type(8)))  float    v8f;

__device__ __forceinline__ void cvt4(v16h& dst, int off, float4 v) {
    dst[off + 0] = (_Float16)v.x;
    dst[off + 1] = (_Float16)v.y;
    dst[off + 2] = (_Float16)v.z;
    dst[off + 3] = (_Float16)v.w;
}

// -- cross-lane add helpers: all intra-row, so DPP/permlane instead of LDS --

template <int CTRL>
__device__ __forceinline__ float xadd_dpp16(float v) {   // v + lane(v, lane^1 or ^2)
#if __has_builtin(__builtin_amdgcn_update_dpp)
    int r = __builtin_amdgcn_update_dpp(0, __float_as_int(v), CTRL, 0xf, 0xf, true);
    return v + __int_as_float(r);
#else
    return v + __shfl_xor(v, (CTRL == 0xB1) ? 1 : 2, 32);
#endif
}

__device__ __forceinline__ float xadd_xor4(float v) {    // v + lane(v, lane^4)
#if __has_builtin(__builtin_amdgcn_mov_dpp8)
    constexpr int sel = 4 | (5 << 3) | (6 << 6) | (7 << 9) |
                        (0 << 12) | (1 << 15) | (2 << 18) | (3 << 21);
    int r = __builtin_amdgcn_mov_dpp8(__float_as_int(v), sel);
    return v + __int_as_float(r);
#else
    return v + __shfl_xor(v, 4, 32);
#endif
}

__device__ __forceinline__ float xadd_xor8(float v) {    // v + lane(v, lane^8)
#if __has_builtin(__builtin_amdgcn_permlane16)
    int r = __builtin_amdgcn_permlane16(__float_as_int(v), __float_as_int(v),
                                        0xFEDCBA98u, 0x76543210u, false, false);
    return v + __int_as_float(r);
#else
    return v + __shfl_xor(v, 8, 32);
#endif
}

// d = (mask bit of this lane) ? b : a  -- one v_cndmask against an SGPR constant
__device__ __forceinline__ float selmask(unsigned mask, float a, float b) {
    float d;
    asm("v_cndmask_b32_e64 %0, %1, %2, %3" : "=v"(d) : "v"(a), "v"(b), "s"(mask));
    return d;
}

__global__ __launch_bounds__(kWavesPerBlock * 32)
void opl_wmma_kernel(const float* __restrict__ x,   // (8192, 39, 32)
                     const float* __restrict__ w,   // (32, 741, 32)  [a, p, c]
                     float* __restrict__ out)       // (8192, 741)
{
    const int lane = threadIdx.x & 31;
    const int wave = threadIdx.x >> 5;
    const unsigned u = blockIdx.x * kWavesPerBlock + wave;   // global wave-unit id
    const int p  = u / kUnitsPerPair;                        // pair (uniform in block)
    const int ub = u % kUnitsPerPair;

    // decode pair index p -> (i = ROW, j = COL)
    int i = 0, base = 0;
    while (base + (kF - 1 - i) <= p) { base += kF - 1 - i; ++i; }
    const int j = i + 1 + (p - base);

    const bool hi  = lane >= 16;
    const int  l16 = lane & 15;
    const bool writer = (lane & 8) == 0;   // lanes 0-7 / 16-23 hold the unique sums

    // ---- B fragments: W_p as B[K=c, N=a], a split into [0,16) and [16,32) ----
    // lane<16: n=lane, c=0..15 ; lane>=16: n=lane-16, c=16..31  (contiguous in c)
    const int aB = l16;
    const int cB = hi ? 16 : 0;
    const float* wp0 = w + ((size_t)aB        * kP + p) * kK + cB;
    const float* wp1 = w + ((size_t)(aB + 16) * kP + p) * kK + cB;
    v16h bf0, bf1;
    {
        const float4* s0 = reinterpret_cast<const float4*>(wp0);
        const float4* s1v = reinterpret_cast<const float4*>(wp1);
#pragma unroll
        for (int t = 0; t < 4; ++t) {
            cvt4(bf0, 4 * t, s0[t]);
            cvt4(bf1, 4 * t, s1v[t]);
        }
    }

    // A-fragment addressing (16-bit A 16x32): lane<16: m=lane, K 0-15 ;
    // lane>=16: m=lane-16, K 16-31
    const int mA    = l16;
    const int cA    = hi ? 16 : 0;
    const int mBase = hi ? 8 : 0;    // C layout: VGPR r -> M=r (low) / r+8 (high)
    const int mOut  = mBase + (lane & 7);

#pragma unroll 2
    for (int t = 0; t < kTilesPerWave; ++t) {
        const int b0 = (ub * kTilesPerWave + t) * 16;

        // ---- A fragment: X[b0+m, i, :], f32 -> f16 ----
        const float* xr = x + ((size_t)(b0 + mA) * kF + i) * kK + cA;
        v16h af;
        cvt4(af, 0,  *reinterpret_cast<const float4*>(xr));
        cvt4(af, 4,  *reinterpret_cast<const float4*>(xr + 4));
        cvt4(af, 8,  *reinterpret_cast<const float4*>(xr + 8));
        cvt4(af, 12, *reinterpret_cast<const float4*>(xr + 12));

        // ---- q gather (one load clause; latency hidden under the WMMAs) ----
        const float* qbase = x + ((size_t)b0 * kF + j) * kK;
        float q0[8], q1[8];
#pragma unroll
        for (int r = 0; r < 8; ++r) {
            const float* qr = qbase + (size_t)(mBase + r) * (kF * kK);
            q0[r] = qr[l16];
            q1[r] = qr[l16 + 16];
        }

        // ---- T = A x B (fp32 accumulate): two WMMAs cover a = 0..31 ----
        v8f c0 = {}, c1 = {};
        c0 = __builtin_amdgcn_wmma_f32_16x16x32_f16(
                 false, af, false, bf0, (short)0, c0, false, false);
        c1 = __builtin_amdgcn_wmma_f32_16x16x32_f16(
                 false, af, false, bf1, (short)0, c1, false, false);

        // ---- u[m, n=l16] = T[m,l16]*q[m,l16] + T[m,l16+16]*q[m,l16+16] ----
        float v0[8];
#pragma unroll
        for (int r = 0; r < 8; ++r) v0[r] = c0[r] * q0[r] + c1[r] * q1[r];

        // ---- DS-free reduction tree over each 16-lane half ----
        // stage 1: xor1 (DPP quad_perm [1,0,3,2]), merge reg pairs by lane bit0
        float s1[8];
#pragma unroll
        for (int r = 0; r < 8; ++r) s1[r] = xadd_dpp16<0xB1>(v0[r]);
        float m1[4];
#pragma unroll
        for (int r = 0; r < 4; ++r) m1[r] = selmask(0xAAAAAAAAu, s1[2 * r], s1[2 * r + 1]);

        // stage 2: xor2 (DPP quad_perm [2,3,0,1]), merge by bit1
        float s2[4];
#pragma unroll
        for (int r = 0; r < 4; ++r) s2[r] = xadd_dpp16<0x4E>(m1[r]);
        float m2[2];
#pragma unroll
        for (int r = 0; r < 2; ++r) m2[r] = selmask(0xCCCCCCCCu, s2[2 * r], s2[2 * r + 1]);

        // stage 3: xor4 (DPP8), merge by bit2
        float s3[2];
#pragma unroll
        for (int r = 0; r < 2; ++r) s3[r] = xadd_xor4(m2[r]);
        float m3 = selmask(0xF0F0F0F0u, s3[0], s3[1]);

        // stage 4: xor8 (permlane16) completes the sum; lane l holds m = mBase+(l&7)
        float total = xadd_xor8(m3);

        if (writer) out[(size_t)(b0 + mOut) * kP + p] = total;
    }
}

extern "C" void kernel_launch(void* const* d_in, const int* in_sizes, int n_in,
                              void* d_out, int out_size, void* d_ws, size_t ws_size,
                              hipStream_t stream) {
    const float* x = (const float*)d_in[0];   // inputs (8192, 39, 32) fp32
    const float* w = (const float*)d_in[1];   // w      (32, 741, 32)  fp32
    float* out = (float*)d_out;               // (8192, 741) fp32

    const int grid = kP * kUnitsPerPair / kWavesPerBlock;   // 741*64/8 = 5928
    opl_wmma_kernel<<<grid, kWavesPerBlock * 32, 0, stream>>>(x, w, out);
}
